// QKRotationWrapper_23957327577945
// MI455X (gfx1250) — matrix operand order repaired and checked by reading, not compiled, not run on hardware
//
#include <hip/hip_runtime.h>

typedef __attribute__((ext_vector_type(16))) _Float16 v16h;
typedef __attribute__((ext_vector_type(8)))  float    v8f;

#define HSCALE 0.08838834764831845f   // 1/sqrt(128)
#define MAXQF  7.0f

// Load one 16x128 f32 tile row-chunk into the four 16x32 f16 A-fragments in
// contiguous-K layout.  Lanes 0-15 hold M=lane, halves 0..15 = K 0..15
// (lanes 16-31 shifted by +16 in K).  So each lane
// needs one contiguous 16-float (64 B) run per fragment -> four float4 loads.
static __device__ __forceinline__ void load_a_frags(const float* __restrict__ row,
                                                    v16h A[4]) {
#pragma unroll
  for (int kk = 0; kk < 4; ++kk) {
    const float4 c0 = *(const float4*)(row + 32 * kk + 0);
    const float4 c1 = *(const float4*)(row + 32 * kk + 4);
    const float4 c2 = *(const float4*)(row + 32 * kk + 8);
    const float4 c3 = *(const float4*)(row + 32 * kk + 12);
    A[kk][0]  = (_Float16)c0.x;  A[kk][1]  = (_Float16)c0.y;
    A[kk][2]  = (_Float16)c0.z;  A[kk][3]  = (_Float16)c0.w;
    A[kk][4]  = (_Float16)c1.x;  A[kk][5]  = (_Float16)c1.y;
    A[kk][6]  = (_Float16)c1.z;  A[kk][7]  = (_Float16)c1.w;
    A[kk][8]  = (_Float16)c2.x;  A[kk][9]  = (_Float16)c2.y;
    A[kk][10] = (_Float16)c2.z;  A[kk][11] = (_Float16)c2.w;
    A[kk][12] = (_Float16)c3.x;  A[kk][13] = (_Float16)c3.y;
    A[kk][14] = (_Float16)c3.z;  A[kk][15] = (_Float16)c3.w;
  }
}

__global__ __launch_bounds__(256) void qk_hadamard_quant_kernel(
    const float* __restrict__ q, const float* __restrict__ k,
    float* __restrict__ oq, float* __restrict__ ok, int ntiles) {
  const int lane = threadIdx.x & 31;
  const int wave = threadIdx.x >> 5;
  const int tile = blockIdx.x * 8 + wave;
  if (tile >= ntiles) return;

  const int nl   = lane & 15;   // column-within-tile / row M for A
  const int half = lane >> 4;

  // Prefetch the K tile (8 KB) while we crunch Q.
  {
    const float* kt = k + (size_t)tile * 2048 + (size_t)lane * 64;
    __builtin_prefetch(kt, 0, 3);
    __builtin_prefetch(kt + 32, 0, 3);
  }

  // ---- Four resident Hadamard B fragments (exact +/-1 in f16) -------------
  // B element e of lane: K = 32*kk + 16*half + e, N = 16*j + nl.
  // sign = parity(e&nl) ^ (half & (j&1)) ^ parity(kk & (j>>1))
  v16h Bp_e, Bn_e, Bp_o, Bn_o;
#pragma unroll
  for (int e = 0; e < 16; ++e) {
    const int be = __popc(e & nl) & 1;   // even-j variant sign bit
    const int bo = be ^ half;            // odd-j variant sign bit
    Bp_e[e] = be ? (_Float16)-1.0f : (_Float16)1.0f;
    Bn_e[e] = be ? (_Float16)1.0f : (_Float16)-1.0f;
    Bp_o[e] = bo ? (_Float16)-1.0f : (_Float16)1.0f;
    Bn_o[e] = bo ? (_Float16)1.0f : (_Float16)-1.0f;
  }

  const size_t row_off = ((size_t)tile * 16 + nl) * 128 + half * 16;
  v16h A[4];

  // ======================= Q: rotate + scale + store =======================
  load_a_frags(q + row_off, A);
  {
    float* o = oq + ((size_t)tile * 16 + half * 8) * 128 + nl;
#pragma unroll
    for (int j = 0; j < 8; ++j) {
      v8f acc = {};
#pragma unroll
      for (int kk = 0; kk < 4; ++kk) {
        const int neg = __popc((unsigned)(kk & (j >> 1))) & 1;  // constant-folded
        v16h B;
        if (j & 1) B = neg ? Bn_o : Bp_o;
        else       B = neg ? Bn_e : Bp_e;
        acc = __builtin_amdgcn_wmma_f32_16x16x32_f16(
            false, A[kk], false, B, (short)0, acc, false, false);
      }
#pragma unroll
      for (int r = 0; r < 8; ++r) o[(size_t)r * 128 + j * 16] = acc[r] * HSCALE;
    }
  }

  // ================= K: rotate, per-row 4-bit fake-quant ===================
  load_a_frags(k + row_off, A);
  v8f acck[8];
#pragma unroll
  for (int j = 0; j < 8; ++j) {
    v8f acc = {};
#pragma unroll
    for (int kk = 0; kk < 4; ++kk) {
      const int neg = __popc((unsigned)(kk & (j >> 1))) & 1;
      v16h B;
      if (j & 1) B = neg ? Bn_o : Bp_o;
      else       B = neg ? Bn_e : Bp_e;
      acc = __builtin_amdgcn_wmma_f32_16x16x32_f16(
          false, A[kk], false, B, (short)0, acc, false, false);
    }
    acck[j] = acc;
  }

  // Row max of |unscaled rotated k|: lane-local over j, then across the 16
  // lanes of this half (same rows m = r + 8*half, columns nl = 0..15).
  float pm[8];
#pragma unroll
  for (int r = 0; r < 8; ++r) {
    float m = fabsf(acck[0][r]);
#pragma unroll
    for (int j = 1; j < 8; ++j) m = fmaxf(m, fabsf(acck[j][r]));
    pm[r] = m;
  }
#pragma unroll
  for (int msk = 1; msk < 16; msk <<= 1) {
#pragma unroll
    for (int r = 0; r < 8; ++r)
      pm[r] = fmaxf(pm[r], __shfl_xor(pm[r], msk, 32));
  }

  // HSCALE cancels inside round(): quantize the unscaled accumulator, apply
  // HSCALE in the dequant scale.  Zero rows -> 0 output (match reference).
  float inv[8], qsc[8];
#pragma unroll
  for (int r = 0; r < 8; ++r) {
    const float xm = (pm[r] == 0.0f) ? 1.0f : pm[r];
    inv[r] = MAXQF / xm;
    qsc[r] = xm * (HSCALE / MAXQF);
  }

  float* o = ok + ((size_t)tile * 16 + half * 8) * 128 + nl;
#pragma unroll
  for (int j = 0; j < 8; ++j) {
#pragma unroll
    for (int r = 0; r < 8; ++r) {
      const float qv =
          fminf(fmaxf(rintf(acck[j][r] * inv[r]), -8.0f), 7.0f);  // v_rndne
      o[(size_t)r * 128 + j * 16] = qv * qsc[r];
    }
  }
}

extern "C" void kernel_launch(void* const* d_in, const int* in_sizes, int n_in,
                              void* d_out, int out_size, void* d_ws, size_t ws_size,
                              hipStream_t stream) {
  (void)n_in; (void)out_size; (void)d_ws; (void)ws_size;
  const float* q = (const float*)d_in[0];
  const float* k = (const float*)d_in[1];
  float* out = (float*)d_out;

  const long long nq    = in_sizes[0];          // elements in q (== k)
  const long long rows  = nq / 128;             // 128-dim head vectors
  const int ntiles      = (int)(rows / 16);     // 16-row tiles, one per wave
  const int blocks      = (ntiles + 7) / 8;     // 8 waves (256 thr) per block

  qk_hadamard_quant_kernel<<<blocks, 256, 0, stream>>>(
      q, k, out, out + nq, ntiles);
}
